// MultiHeadSelfAttention_22960895164731
// MI455X (gfx1250) — matrix operand order repaired and checked by reading, not compiled, not run on hardware
//
#include <hip/hip_runtime.h>
#include <hip/hip_bf16.h>

// ---------------------------------------------------------------------------
// MI455X (gfx1250) fused multi-head self-attention.
//   B=2, S=2048, D=1024, H=16, hd=64.
//   All matmuls via v_wmma_f32_16x16x32_bf16 (wave32, fp32 accum).
//   GEMM tiles staged LDS-side by the Tensor Data Mover (tensor_load_to_lds,
//   double-buffered, s_wait_tensorcnt), fragments read as ds_load_b128.
//   fp32->bf16 conversion hoisted into bandwidth-bound prologue kernels.
// ---------------------------------------------------------------------------

typedef __attribute__((ext_vector_type(16))) __bf16 v16bf;
typedef __attribute__((ext_vector_type(8)))  float  v8f;
typedef unsigned int __attribute__((ext_vector_type(4))) v4ui;
typedef int          __attribute__((ext_vector_type(4))) v4si;
typedef int          __attribute__((ext_vector_type(8))) v8si;

#define WMMA_BF16(a, b, c) \
  __builtin_amdgcn_wmma_f32_16x16x32_bf16(false, (a), false, (b), (short)0, (c), false, false)

union AF {
  v16bf          v;
  unsigned       u[8];
  uint4          q[2];
  __bf16         b[16];
};

static __device__ __forceinline__ __bf16 f2bf(float f) {
  union { float f; unsigned u; } a; a.f = f;
  unsigned r = a.u + 0x7FFFu + ((a.u >> 16) & 1u);   // round-to-nearest-even
  union { unsigned short s; __bf16 b; } o; o.s = (unsigned short)(r >> 16);
  return o.b;
}

// ---------------------------------------------------------------------------
// TDM 2-D tile load: global (row-major, elem stride0 between rows, bf16) ->
// LDS (packed tile rows).  D# per CDNA5 ISA.
//   group0: [1:0]=count=1 | lds_addr | global_addr[56:0] | type=2
//   group1: data_size=1(2B) | tensor_dim0/1 | tile_dim0/1 | dim0_stride
//   groups 2/3: zero (tile_dim3/4 = 0 => 2-D)
// ---------------------------------------------------------------------------
static __device__ __forceinline__ void tdm_load_2d(
    unsigned lds_off, const void* gptr,
    unsigned tdim0, unsigned tdim1, unsigned tile0, unsigned tile1,
    unsigned stride0) {
  const unsigned long long ga = (unsigned long long)(uintptr_t)gptr;
  v4ui g0;
  g0[0] = 1u;                                            // count=1, user mode
  g0[1] = lds_off;                                       // LDS byte address
  g0[2] = (unsigned)ga;                                  // global_addr[31:0]
  g0[3] = (unsigned)((ga >> 32) & 0x1FFFFFFu) | (2u << 30); // [56:32] | type=2
  v8si g1;
  g1[0] = (int)(1u << 16);                               // data_size = 2 bytes
  g1[1] = (int)((tdim0 & 0xFFFFu) << 16);                // tensor_dim0[15:0]
  g1[2] = (int)((tdim0 >> 16) | ((tdim1 & 0xFFFFu) << 16));
  g1[3] = (int)((tdim1 >> 16) | (tile0 << 16));          // tile_dim0
  g1[4] = (int)tile1;                                    // tile_dim1 (tile_dim2=0)
  g1[5] = (int)stride0;                                  // dim0 stride [31:0]
  g1[6] = 0;
  g1[7] = 0;
  v4si z4 = {0, 0, 0, 0};
#if defined(__clang_major__) && (__clang_major__ >= 23)
  v8si z8 = {0, 0, 0, 0, 0, 0, 0, 0};
  __builtin_amdgcn_tensor_load_to_lds(g0, g1, z4, z4, z8, 0);
#else
  __builtin_amdgcn_tensor_load_to_lds(g0, g1, z4, z4, 0);
#endif
}

// Fragment layouts (CDNA5 ISA):
//  A (16x32 MxK):  row M = lane&15; K = {0..7,16..23} / {8..15,24..31} by half.
//  B (32x16 KxN):  col N = lane&15; K = {0..7,16..23} / {8..15,24..31} by half.
//  C/D (16x16 f32): row = r + 8*half, col = lane&15.

// ---------------------------------------------------------------------------
// Prologue A: x fp32 -> bf16, same layout. 4 elements/thread.
// ---------------------------------------------------------------------------
__global__ __launch_bounds__(256)
void convert_bf16_kernel(const float* __restrict__ X, __bf16* __restrict__ Xb) {
  const size_t i = ((size_t)blockIdx.x * 256 + threadIdx.x) * 4;
  float4 f = *(const float4*)(X + i);
  union { __bf16 b[4]; uint2 u; } o;
  o.b[0] = f2bf(f.x); o.b[1] = f2bf(f.y); o.b[2] = f2bf(f.z); o.b[3] = f2bf(f.w);
  *(uint2*)(Xb + i) = o.u;
}

// ---------------------------------------------------------------------------
// Prologue B: W[K,N] fp32 -> Wt[N,K] bf16 (LDS-tiled 32x32 transpose).
// ---------------------------------------------------------------------------
__global__ __launch_bounds__(256)
void convert_transpose_kernel(const float* __restrict__ W, __bf16* __restrict__ Wt,
                              int K, int N) {
  __shared__ float tile[32][33];
  const int n0 = blockIdx.x * 32, k0 = blockIdx.y * 32;
  const int tx = threadIdx.x & 31, ty = threadIdx.x >> 5;   // 32 x 8
#pragma unroll
  for (int i = 0; i < 32; i += 8)
    tile[ty + i][tx] = W[(size_t)(k0 + ty + i) * N + n0 + tx];
  __syncthreads();
#pragma unroll
  for (int i = 0; i < 32; i += 8)
    Wt[(size_t)(n0 + ty + i) * K + k0 + tx] = f2bf(tile[tx][ty + i]);
}

// ---------------------------------------------------------------------------
// Kernel 1: QKV projection.  Xb[4096,1024]bf16 * Wqt[3072,1024]bf16(T).
// Block = 8 waves, 128x64 tile.  Per 32-deep k-step, wave 0 issues two TDM
// loads (A 128x32, B 64x32) into the next LDS buffer while all waves compute
// from the current one.  Results scattered to:
//   Q,K bf16 [B,H,S,64]  (hd contiguous)   V bf16 [B,H,64,S] (key contiguous)
// ---------------------------------------------------------------------------
__global__ __launch_bounds__(256)
void qkv_gemm_kernel(const __bf16* __restrict__ Xb, const __bf16* __restrict__ Wqt,
                     __bf16* __restrict__ Qb, __bf16* __restrict__ Kb,
                     __bf16* __restrict__ Vt) {
  __shared__ __align__(16) __bf16 At[2][128 * 32];   // 16 KB
  __shared__ __align__(16) __bf16 Bt[2][64 * 32];    //  8 KB
  const int lane = threadIdx.x & 31;
  const int w    = threadIdx.x >> 5;
  const int lh   = lane >> 4;
  const int ln   = lane & 15;
  const int Mblk = blockIdx.y * 128, Nblk = blockIdx.x * 64;
  const int wm   = (w >> 1) * 32,    wn   = (w & 1) * 32;

  v8f acc[2][2] = {};
  if (w == 0) {
    tdm_load_2d((unsigned)(uintptr_t)&At[0][0], Xb + (size_t)Mblk * 1024,
                1024, 4096, 32, 128, 1024);
    tdm_load_2d((unsigned)(uintptr_t)&Bt[0][0], Wqt + (size_t)Nblk * 1024,
                1024, 3072, 32, 64, 1024);
  }

  for (int it = 0; it < 32; ++it) {
    const int cur = it & 1;
    if (w == 0) {
      if (it < 31) {
        const int kn = (it + 1) * 32;
        tdm_load_2d((unsigned)(uintptr_t)&At[cur ^ 1][0],
                    Xb + (size_t)Mblk * 1024 + kn, 1024, 4096, 32, 128, 1024);
        tdm_load_2d((unsigned)(uintptr_t)&Bt[cur ^ 1][0],
                    Wqt + (size_t)Nblk * 1024 + kn, 1024, 3072, 32, 64, 1024);
        __builtin_amdgcn_s_wait_tensorcnt(2);   // current buffer complete
      } else {
        __builtin_amdgcn_s_wait_tensorcnt(0);
      }
    }
    __syncthreads();                            // buffer `cur` visible to all

    AF af[2], bf[2];
#pragma unroll
    for (int mi = 0; mi < 2; ++mi) {
      const __bf16* ap = &At[cur][(wm + mi * 16 + ln) * 32 + 8 * lh];
      af[mi].q[0] = *(const uint4*)(ap);        // K 0..7  / 8..15
      af[mi].q[1] = *(const uint4*)(ap + 16);   // K 16..23 / 24..31
    }
#pragma unroll
    for (int ni = 0; ni < 2; ++ni) {
      const __bf16* bp = &Bt[cur][(wn + ni * 16 + ln) * 32 + 8 * lh];
      bf[ni].q[0] = *(const uint4*)(bp);        // K 0..7  / 8..15
      bf[ni].q[1] = *(const uint4*)(bp + 16);   // K 16..23 / 24..31
    }
#pragma unroll
    for (int mi = 0; mi < 2; ++mi)
#pragma unroll
      for (int ni = 0; ni < 2; ++ni)
        acc[mi][ni] = WMMA_BF16(af[mi].v, bf[ni].v, acc[mi][ni]);

    __syncthreads();                            // reads done before next DMA
  }

  // Scatter into Q/K [B,H,S,64] and V [B,H,64,S] (bf16).
#pragma unroll
  for (int mi = 0; mi < 2; ++mi) {
#pragma unroll
    for (int ni = 0; ni < 2; ++ni) {
      const int ebase = Nblk + wn + ni * 16;       // uniform per tile
      const int which = ebase >> 10;               // 0=Q 1=K 2=V
      const int h     = (ebase & 1023) >> 6;
#pragma unroll
      for (int r = 0; r < 8; ++r) {
        const int g   = Mblk + wm + mi * 16 + r + 8 * lh;  // b*2048+s
        const int b   = g >> 11, s = g & 2047;
        const int hdi = (ebase & 63) + ln;
        const __bf16 val = f2bf(acc[mi][ni][r]);
        if (which == 2)
          Vt[((size_t)(b * 16 + h) * 64 + hdi) * 2048 + s] = val;
        else if (which == 1)
          Kb[((size_t)(b * 16 + h) * 2048 + s) * 64 + hdi] = val;
        else
          Qb[((size_t)(b * 16 + h) * 2048 + s) * 64 + hdi] = val;
      }
    }
  }
}

// ---------------------------------------------------------------------------
// Kernel 2: causal flash attention (unscaled scores, exact -1e9 fill, per
// reference).  One wave per 16-row query tile; key blocks of 32; softmax rows
// reduced with __shfl_xor within 16-lane halves; P transposed C->A layout
// through a per-wave 16x32 bf16 LDS tile fenced with s_wait_dscnt.
// Output merged heads [B,S,D] bf16.
// ---------------------------------------------------------------------------
__global__ __launch_bounds__(128)
void attn_kernel(const __bf16* __restrict__ Qb, const __bf16* __restrict__ Kb,
                 const __bf16* __restrict__ Vt, __bf16* __restrict__ Mg) {
  __shared__ __align__(16) __bf16 plds[4][16 * 32];
  const int lane  = threadIdx.x & 31;
  const int wv    = threadIdx.x >> 5;
  const int lh    = lane >> 4;
  const int ln    = lane & 15;
  const int bh    = blockIdx.x;                 // 0..31 = b*16+h
  const int qt    = blockIdx.y * 4 + wv;        // 0..127
  const int qbase = qt * 16;
  __bf16* pl = plds[wv];

  // Q A-fragments for hd 0..31 and 32..63 (loaded once).
  const __bf16* Qrow = Qb + ((size_t)bh * 2048 + qbase + ln) * 64;
  AF qf[2];
#pragma unroll
  for (int hb = 0; hb < 2; ++hb) {
    qf[hb].q[0] = *(const uint4*)(Qrow + hb * 32 + 8 * lh);
    qf[hb].q[1] = *(const uint4*)(Qrow + hb * 32 + 8 * lh + 16);
  }

  float m[8], l[8], scale[8];
#pragma unroll
  for (int r = 0; r < 8; ++r) { m[r] = -3.0e38f; l[r] = 0.0f; }
  v8f o[4] = {};

  for (int kb = 0; kb < qbase + 16; kb += 32) {
    // scores for key sub-tiles c=0 (keys kb..kb+15), c=1 (kb+16..kb+31)
    v8f s[2] = {};
#pragma unroll
    for (int c = 0; c < 2; ++c) {
      const __bf16* Kp = Kb + ((size_t)bh * 2048 + kb + c * 16 + ln) * 64 + 8 * lh;
#pragma unroll
      for (int hb = 0; hb < 2; ++hb) {
        AF kf;
        kf.q[0] = *(const uint4*)(Kp + hb * 32);
        kf.q[1] = *(const uint4*)(Kp + hb * 32 + 16);
        s[c] = WMMA_BF16(qf[hb].v, kf.v, s[c]);
      }
    }

    // Online softmax: row M = r + 8*lh lives within one 16-lane half.
#pragma unroll
    for (int r = 0; r < 8; ++r) {
      const int   qg  = qbase + r + 8 * lh;
      const int   kg0 = kb + ln, kg1 = kb + 16 + ln;
      const float a0  = (kg0 > qg) ? -1.0e9f : s[0][r];
      const float a1  = (kg1 > qg) ? -1.0e9f : s[1][r];
      float mx = fmaxf(a0, a1);
#pragma unroll
      for (int off = 1; off < 16; off <<= 1)
        mx = fmaxf(mx, __shfl_xor(mx, off, 32));
      const float mn = fmaxf(m[r], mx);
      const float sc = __expf(m[r] - mn);
      const float p0 = (kg0 > qg) ? 0.0f : __expf(a0 - mn);
      const float p1 = (kg1 > qg) ? 0.0f : __expf(a1 - mn);
      float rs = p0 + p1;
#pragma unroll
      for (int off = 1; off < 16; off <<= 1)
        rs += __shfl_xor(rs, off, 32);
      l[r]     = l[r] * sc + rs;
      m[r]     = mn;
      scale[r] = sc;
      // stage P (C layout) into LDS row-major 16x32 for transpose
      pl[(r + 8 * lh) * 32 + ln]      = f2bf(p0);
      pl[(r + 8 * lh) * 32 + 16 + ln] = f2bf(p1);
    }

    // rescale running output accumulators
#pragma unroll
    for (int t = 0; t < 4; ++t)
#pragma unroll
      for (int r = 0; r < 8; ++r)
        o[t][r] *= scale[r];

    // per-wave LDS RAW fence (CDNA5 split counter; LDS ops in-order per wave)
    asm volatile("s_wait_dscnt 0" ::: "memory");

    // re-load P as an A-fragment (row = ln, K per A layout)
    AF pf;
    {
      const __bf16* pp = pl + ln * 32 + 8 * lh;
      pf.q[0] = *(const uint4*)pp;         // K 0..7  / 8..15
      pf.q[1] = *(const uint4*)(pp + 16);  // K 16..23 / 24..31
    }

    // P(16x32) x V(32x64): V transposed [B,H,64,S] so key dim is contiguous
#pragma unroll
    for (int t = 0; t < 4; ++t) {
      AF vf;
      const __bf16* Vp = Vt + ((size_t)bh * 64 + t * 16 + ln) * 2048 + kb + 8 * lh;
      vf.q[0] = *(const uint4*)(Vp);       // keys 0..7  / 8..15
      vf.q[1] = *(const uint4*)(Vp + 16);  // keys 16..23 / 24..31
      o[t] = WMMA_BF16(pf.v, vf.v, o[t]);
    }
  }

  // finalize: divide by row sums, write merged [B,S,H*64] bf16
  const int b = bh >> 4, h = bh & 15;
#pragma unroll
  for (int r = 0; r < 8; ++r) {
    const float inv  = 1.0f / l[r];
    const int   srow = qbase + r + 8 * lh;
    __bf16* mp = Mg + ((size_t)(b * 2048 + srow)) * 1024 + h * 64;
#pragma unroll
    for (int t = 0; t < 4; ++t)
      mp[t * 16 + ln] = f2bf(o[t][r] * inv);
  }
}

// ---------------------------------------------------------------------------
// Kernel 3: output projection.  Mg[4096,1024]bf16 * Wot[1024,1024]bf16(T)
// -> out fp32 [4096,1024].  Same TDM-staged structure as kernel 1.
// ---------------------------------------------------------------------------
__global__ __launch_bounds__(256)
void out_gemm_kernel(const __bf16* __restrict__ A, const __bf16* __restrict__ Wot,
                     float* __restrict__ Out) {
  __shared__ __align__(16) __bf16 At[2][128 * 32];
  __shared__ __align__(16) __bf16 Bt[2][64 * 32];
  const int lane = threadIdx.x & 31;
  const int w    = threadIdx.x >> 5;
  const int lh   = lane >> 4;
  const int ln   = lane & 15;
  const int Mblk = blockIdx.y * 128, Nblk = blockIdx.x * 64;
  const int wm   = (w >> 1) * 32,    wn   = (w & 1) * 32;

  v8f acc[2][2] = {};
  if (w == 0) {
    tdm_load_2d((unsigned)(uintptr_t)&At[0][0], A + (size_t)Mblk * 1024,
                1024, 4096, 32, 128, 1024);
    tdm_load_2d((unsigned)(uintptr_t)&Bt[0][0], Wot + (size_t)Nblk * 1024,
                1024, 1024, 32, 64, 1024);
  }

  for (int it = 0; it < 32; ++it) {
    const int cur = it & 1;
    if (w == 0) {
      if (it < 31) {
        const int kn = (it + 1) * 32;
        tdm_load_2d((unsigned)(uintptr_t)&At[cur ^ 1][0],
                    A + (size_t)Mblk * 1024 + kn, 1024, 4096, 32, 128, 1024);
        tdm_load_2d((unsigned)(uintptr_t)&Bt[cur ^ 1][0],
                    Wot + (size_t)Nblk * 1024 + kn, 1024, 1024, 32, 64, 1024);
        __builtin_amdgcn_s_wait_tensorcnt(2);
      } else {
        __builtin_amdgcn_s_wait_tensorcnt(0);
      }
    }
    __syncthreads();

    AF af[2], bf[2];
#pragma unroll
    for (int mi = 0; mi < 2; ++mi) {
      const __bf16* ap = &At[cur][(wm + mi * 16 + ln) * 32 + 8 * lh];
      af[mi].q[0] = *(const uint4*)(ap);
      af[mi].q[1] = *(const uint4*)(ap + 16);
    }
#pragma unroll
    for (int ni = 0; ni < 2; ++ni) {
      const __bf16* bp = &Bt[cur][(wn + ni * 16 + ln) * 32 + 8 * lh];
      bf[ni].q[0] = *(const uint4*)(bp);
      bf[ni].q[1] = *(const uint4*)(bp + 16);
    }
#pragma unroll
    for (int mi = 0; mi < 2; ++mi)
#pragma unroll
      for (int ni = 0; ni < 2; ++ni)
        acc[mi][ni] = WMMA_BF16(af[mi].v, bf[ni].v, acc[mi][ni]);

    __syncthreads();
  }

#pragma unroll
  for (int mi = 0; mi < 2; ++mi)
#pragma unroll
    for (int ni = 0; ni < 2; ++ni)
#pragma unroll
      for (int r = 0; r < 8; ++r)
        Out[(size_t)(Mblk + wm + mi * 16 + r + 8 * lh) * 1024 +
            Nblk + wn + ni * 16 + ln] = acc[mi][ni][r];
}

// ---------------------------------------------------------------------------
extern "C" void kernel_launch(void* const* d_in, const int* in_sizes, int n_in,
                              void* d_out, int out_size, void* d_ws, size_t ws_size,
                              hipStream_t stream) {
  const float* x    = (const float*)d_in[0];  // [2,2048,1024]
  const float* wqkv = (const float*)d_in[1];  // [1024,3072]
  const float* wout = (const float*)d_in[2];  // [1024,1024]
  float*       out  = (float*)d_out;          // [2,2048,1024]

  char* ws = (char*)d_ws;                     // needs 48 MB
  const size_t MB = (size_t)1 << 20;
  __bf16* Xb  = (__bf16*)(ws);                 // 8 MB  [4096,1024]
  __bf16* Wqt = (__bf16*)(ws + 8 * MB);        // 6 MB  [3072,1024] (transposed)
  __bf16* Wot = (__bf16*)(ws + 14 * MB);       // 2 MB  [1024,1024] (transposed)
  __bf16* Qb  = (__bf16*)(ws + 16 * MB);       // 8 MB  [B,H,S,64]
  __bf16* Kb  = (__bf16*)(ws + 24 * MB);       // 8 MB  [B,H,S,64]
  __bf16* Vt  = (__bf16*)(ws + 32 * MB);       // 8 MB  [B,H,64,S]
  __bf16* Mg  = (__bf16*)(ws + 40 * MB);       // 8 MB  merged heads

  convert_bf16_kernel<<<4096, 256, 0, stream>>>(x, Xb);
  convert_transpose_kernel<<<dim3(96, 32), 256, 0, stream>>>(wqkv, Wqt, 1024, 3072);
  convert_transpose_kernel<<<dim3(32, 32), 256, 0, stream>>>(wout, Wot, 1024, 1024);
  qkv_gemm_kernel<<<dim3(48, 32), 256, 0, stream>>>(Xb, Wqt, Qb, Kb, Vt);
  attn_kernel<<<dim3(32, 32), 128, 0, stream>>>(Qb, Kb, Vt, Mg);
  out_gemm_kernel<<<dim3(16, 32), 256, 0, stream>>>(Mg, Wot, out);
}
